// ConvBertSelfAttention_55327768707586
// MI455X (gfx1250) — matrix operand order repaired and checked by reading, not compiled, not run on hardware
//
#include <hip/hip_runtime.h>
#include <hip/hip_bf16.h>

typedef __bf16 bf16_t;
typedef __attribute__((ext_vector_type(16))) __bf16 v16bf;
typedef __attribute__((ext_vector_type(8)))  float  v8f;

#define HIDDEN 768
#define AH     384
#define NH     6
#define HD     64
#define KW     9
#define BATCH  4
#define SEQ    2048
#define NTOK   (BATCH * SEQ)

struct __align__(16) U4 { unsigned int x, y, z, w; };

// ---------- CDNA5 async global->LDS copy (ASYNCcnt-tracked, ISA §10.7/§15.18) ----------
// Copies 16 bytes per lane directly into LDS without touching VGPRs/VALU.
__device__ inline void async_cp16(void* ldsdst, const void* gsrc) {
  unsigned l = (unsigned)(uintptr_t)ldsdst;            // LDS aperture: low 32 bits = LDS offset
  unsigned long long g = (unsigned long long)(uintptr_t)gsrc;
  asm volatile("global_load_async_to_lds_b128 %0, %1, off" :: "v"(l), "v"(g) : "memory");
}
__device__ inline void wait_async0() {
  asm volatile("s_wait_asynccnt 0" ::: "memory");
}

// ---------- WMMA fragment loaders (CDNA5 wave32, 16-bit layouts, ISA 7.12.2) ----------
// A matrix 16x32 bf16: lane L (M=L%16) holds K = 8*half+(0..7) and 16+8*half+(0..7), half=L>=16
__device__ inline v16bf frag_A(const bf16_t* row, int lane) {
  int half = (lane >> 4) & 1;
  union { U4 u[2]; v16bf v; } c;
  c.u[0] = *(const U4*)(row + half * 8);
  c.u[1] = *(const U4*)(row + 16 + half * 8);
  return c.v;
}
// B matrix 32x16 bf16, stored N-major ([N][K] rows of >=32 contiguous K values):
// lane L (N=L%16) holds K = 8*half+(0..7) and 16+8*half+(0..7), half=L>=16
__device__ inline v16bf frag_B(const bf16_t* row, int lane) {
  int half = (lane >> 4) & 1;
  union { U4 u[2]; v16bf v; } c;
  c.u[0] = *(const U4*)(row + half * 8);
  c.u[1] = *(const U4*)(row + 16 + half * 8);
  return c.v;
}
__device__ inline v8f zero8() {
  v8f z;
#pragma unroll
  for (int i = 0; i < 8; i++) z[i] = 0.f;
  return z;
}

// ---------- small utility kernels ----------
__global__ void cvt_f32_bf16(const float* __restrict__ in, bf16_t* __restrict__ out, int n) {
  int i = blockIdx.x * blockDim.x + threadIdx.x;
  if (i < n) out[i] = (bf16_t)in[i];
}

// W [Kd, N] row-major fp32 -> Wt [N][Kd] bf16
__global__ void tpack(const float* __restrict__ W, bf16_t* __restrict__ Wt, int Kd, int N) {
  int i = blockIdx.x * blockDim.x + threadIdx.x;
  if (i >= Kd * N) return;
  int n = i / Kd, kcol = i % Kd;
  Wt[i] = (bf16_t)W[(size_t)kcol * N + n];
}

// depthwise conv over seq (per batch, zero pad), output bf16 [token][c]
__global__ void dwconv(const float* __restrict__ x, const float* __restrict__ wk,
                       bf16_t* __restrict__ dwb) {
  int i = blockIdx.x * blockDim.x + threadIdx.x;
  if (i >= NTOK * HIDDEN) return;
  int c = i % HIDDEN;
  int token = i / HIDDEN;
  int b = token / SEQ, s = token % SEQ;
  float acc = 0.f;
#pragma unroll
  for (int j = 0; j < KW; j++) {
    int ss = s + j - (KW - 1) / 2;
    if (ss >= 0 && ss < SEQ)
      acc += x[((size_t)b * SEQ + ss) * HIDDEN + c] * wk[c * KW + j];
  }
  dwb[i] = (bf16_t)acc;
}

__global__ void mul_inplace(float* __restrict__ a, const float* __restrict__ b, int n) {
  int i = blockIdx.x * blockDim.x + threadIdx.x;
  if (i < n) a[i] *= b[i];
}

// dynamic kernel: per (token, head) compute 9 logits = ca_row @ Wck[:, h*9+j] + bck, softmax
__global__ void dynk(const float* __restrict__ ca, const float* __restrict__ Wck,
                     const float* __restrict__ bck, float* __restrict__ kern, int n) {
  int i = blockIdx.x * blockDim.x + threadIdx.x;
  if (i >= n) return;
  int h = i % NH;
  size_t token = i / NH;
  float lg[KW];
#pragma unroll
  for (int j = 0; j < KW; j++) lg[j] = bck[h * KW + j];
  const float* row = ca + token * AH;
  for (int c = 0; c < AH; c++) {
    float x = row[c];
    const float* w = Wck + (size_t)c * (NH * KW) + h * KW;
#pragma unroll
    for (int j = 0; j < KW; j++) lg[j] += x * w[j];
  }
  float mx = lg[0];
#pragma unroll
  for (int j = 1; j < KW; j++) mx = fmaxf(mx, lg[j]);
  float s = 0.f;
#pragma unroll
  for (int j = 0; j < KW; j++) { lg[j] = __expf(lg[j] - mx); s += lg[j]; }
  float inv = 1.f / s;
#pragma unroll
  for (int j = 0; j < KW; j++) kern[(token * NH + h) * KW + j] = lg[j] * inv;
}

// windowed contraction: out[token, 384 + col] = sum_k co[s+k-4, col] * kern[token,h,k]
__global__ void convctx(const float* __restrict__ co, const float* __restrict__ kern,
                        float* __restrict__ out) {
  int i = blockIdx.x * blockDim.x + threadIdx.x;
  if (i >= NTOK * AH) return;
  int col = i % AH;
  size_t token = i / AH;
  int b = (int)(token / SEQ), s = (int)(token % SEQ);
  int h = col >> 6;
  const float* kr = kern + (token * NH + h) * KW;
  float acc = 0.f;
#pragma unroll
  for (int j = 0; j < KW; j++) {
    int ss = s + j - (KW - 1) / 2;
    if (ss >= 0 && ss < SEQ)
      acc += co[((size_t)b * SEQ + ss) * AH + col] * kr[j];
  }
  out[token * (2 * AH) + AH + col] = acc;
}

// ---------- WMMA GEMM: C[M,N] = A[M,Kd] @ Bt[N,Kd]^T + bias ----------
// Double-buffered LDS, tiles filled by async global->LDS DMA overlapped with WMMA.
template <bool WB>
__global__ __launch_bounds__(256) void gemm_bf16(
    const bf16_t* __restrict__ A, const bf16_t* __restrict__ Bt,
    const float* __restrict__ bias, float* __restrict__ C,
    bf16_t* __restrict__ Cb, int M, int N, int Kd) {
  __shared__ __align__(16) bf16_t lA[2][128 * 32];
  __shared__ __align__(16) bf16_t lB[2][128 * 32];
  int tid = threadIdx.x;
  int lane = tid & 31, wave = tid >> 5;
  int lr = lane & 15, half = lane >> 4;
  int wm = (wave >> 1) * 32, wn = (wave & 1) * 64;
  int gM = blockIdx.y * 128, gN = blockIdx.x * 128;
  int row = tid >> 1, hh = (tid & 1) * 16;

  v8f acc[2][4];
#pragma unroll
  for (int i = 0; i < 2; i++)
#pragma unroll
    for (int j = 0; j < 4; j++) acc[i][j] = zero8();

  const bf16_t* gA = A + (size_t)(gM + row) * Kd + hh;
  const bf16_t* gB = Bt + (size_t)(gN + row) * Kd + hh;
  bf16_t* dA = &lA[0][row * 32 + hh];  // buffer stride = 128*32 elements
  bf16_t* dB = &lB[0][row * 32 + hh];
  const int BUFE = 128 * 32;

  // prologue: async-fill buffer 0 (2x16B per thread per tile)
  async_cp16(dA, gA);
  async_cp16(dA + 8, gA + 8);
  async_cp16(dB, gB);
  async_cp16(dB + 8, gB + 8);

  int nk = Kd >> 5;
  for (int i = 0; i < nk; i++) {
    wait_async0();
    __syncthreads();
    if (i + 1 < nk) {  // async-fill next buffer while computing current
      int nb = (i + 1) & 1;
      int kk = (i + 1) * 32;
      async_cp16(dA + nb * BUFE, gA + kk);
      async_cp16(dA + nb * BUFE + 8, gA + kk + 8);
      async_cp16(dB + nb * BUFE, gB + kk);
      async_cp16(dB + nb * BUFE + 8, gB + kk + 8);
    }
    const bf16_t* la = lA[i & 1];
    const bf16_t* lb = lB[i & 1];
    v16bf af[2], bfm[4];
#pragma unroll
    for (int mt = 0; mt < 2; mt++) af[mt] = frag_A(la + (wm + mt * 16 + lr) * 32, lane);
#pragma unroll
    for (int nt = 0; nt < 4; nt++) bfm[nt] = frag_B(lb + (wn + nt * 16 + lr) * 32, lane);
#pragma unroll
    for (int mt = 0; mt < 2; mt++)
#pragma unroll
      for (int nt = 0; nt < 4; nt++)
        acc[mt][nt] = __builtin_amdgcn_wmma_f32_16x16x32_bf16(
            false, af[mt], false, bfm[nt], (short)0, acc[mt][nt], false, false);
  }

#pragma unroll
  for (int mt = 0; mt < 2; mt++) {
    int rbase = gM + wm + mt * 16 + 8 * half;
#pragma unroll
    for (int nt = 0; nt < 4; nt++) {
      int cc = gN + wn + nt * 16 + lr;
      float bv = bias[cc];
      size_t base = (size_t)rbase * N + cc;
#pragma unroll
      for (int r = 0; r < 8; r++) {
        float v = acc[mt][nt][r] + bv;
        C[base + (size_t)r * N] = v;
        if constexpr (WB) Cb[base + (size_t)r * N] = (bf16_t)v;
      }
    }
  }
}

// ---------- flash attention (per b,h,64-q-row block; 4 waves x 16 q rows) ----------
__global__ __launch_bounds__(128) void attn(
    const bf16_t* __restrict__ qb, const bf16_t* __restrict__ kb,
    const bf16_t* __restrict__ vb, const float* __restrict__ mask,
    float* __restrict__ out) {
  __shared__ __align__(16) bf16_t Kt[32 * 64];     // [kcol][d]  (B for QK^T)
  __shared__ __align__(16) bf16_t Vt[64 * 32];     // [d][kidx]  (B for P@V)
  __shared__ __align__(16) bf16_t Pl[4][16 * 32];  // per-wave P staging
  int tid = threadIdx.x, lane = tid & 31, wave = tid >> 5;
  int lr = lane & 15, half = lane >> 4;
  int blk = blockIdx.x;
  int qblk = blk & 31;
  int bh = blk >> 5;
  int h = bh % NH, b = bh / NH;
  int q0 = qblk * 64 + wave * 16;
  const size_t tokbase = (size_t)b * SEQ;

  v16bf aq[2];
#pragma unroll
  for (int dc = 0; dc < 2; dc++)
    aq[dc] = frag_A(qb + (tokbase + q0 + lr) * AH + h * HD + dc * 32, lane);

  float m_r[8], l_r[8];
  v8f o[4];
#pragma unroll
  for (int r = 0; r < 8; r++) { m_r[r] = -3.0e38f; l_r[r] = 0.f; }
#pragma unroll
  for (int dt = 0; dt < 4; dt++) o[dt] = zero8();

  for (int sb = 0; sb < SEQ; sb += 32) {
    {  // K tile [32][64] via async DMA (2x16B per thread)
      int row = tid >> 2, q4 = (tid & 3) * 16;
      const bf16_t* src = kb + (tokbase + sb + row) * AH + h * HD + q4;
      bf16_t* dst = &Kt[row * 64 + q4];
      async_cp16(dst, src);
      async_cp16(dst + 8, src + 8);
    }
    for (int c = tid; c < 256; c += 128) {  // V tile transposed -> [d][kidx]
      int kk = c >> 3, d0 = (c & 7) * 8;
      union { U4 u; bf16_t e[8]; } vv;
      vv.u = *(const U4*)(vb + (tokbase + sb + kk) * AH + h * HD + d0);
#pragma unroll
      for (int j = 0; j < 8; j++) Vt[(d0 + j) * 32 + kk] = vv.e[j];
    }
    wait_async0();
    __syncthreads();

    v8f s0 = zero8(), s1 = zero8();
#pragma unroll
    for (int dc = 0; dc < 2; dc++) {
      s0 = __builtin_amdgcn_wmma_f32_16x16x32_bf16(
          false, aq[dc], false, frag_B(Kt + lr * 64 + dc * 32, lane), (short)0, s0, false, false);
      s1 = __builtin_amdgcn_wmma_f32_16x16x32_bf16(
          false, aq[dc], false, frag_B(Kt + (16 + lr) * 64 + dc * 32, lane), (short)0, s1, false, false);
    }
    float mv0 = mask[tokbase + sb + lr];
    float mv1 = mask[tokbase + sb + 16 + lr];

    float p0[8], p1[8];
#pragma unroll
    for (int r = 0; r < 8; r++) {
      float a = s0[r] * 0.125f + mv0;
      float c = s1[r] * 0.125f + mv1;
      float t = fmaxf(a, c);
      t = fmaxf(t, __shfl_xor(t, 1));
      t = fmaxf(t, __shfl_xor(t, 2));
      t = fmaxf(t, __shfl_xor(t, 4));
      t = fmaxf(t, __shfl_xor(t, 8));
      float mn = fmaxf(m_r[r], t);
      float sc = __expf(m_r[r] - mn);
      p0[r] = __expf(a - mn);
      p1[r] = __expf(c - mn);
      float rs = p0[r] + p1[r];
      rs += __shfl_xor(rs, 1);
      rs += __shfl_xor(rs, 2);
      rs += __shfl_xor(rs, 4);
      rs += __shfl_xor(rs, 8);
      l_r[r] = l_r[r] * sc + rs;
      m_r[r] = mn;
#pragma unroll
      for (int dt = 0; dt < 4; dt++) o[dt][r] *= sc;
    }

    bf16_t* P = Pl[wave];
#pragma unroll
    for (int r = 0; r < 8; r++) {
      P[(r + 8 * half) * 32 + lr]      = (bf16_t)p0[r];
      P[(r + 8 * half) * 32 + 16 + lr] = (bf16_t)p1[r];
    }
    __syncthreads();

    v16bf aP = frag_A(P + lr * 32, lane);
#pragma unroll
    for (int dt = 0; dt < 4; dt++)
      o[dt] = __builtin_amdgcn_wmma_f32_16x16x32_bf16(
          false, aP, false, frag_B(Vt + (dt * 16 + lr) * 32, lane), (short)0, o[dt], false, false);
    __syncthreads();
  }

#pragma unroll
  for (int dt = 0; dt < 4; dt++)
#pragma unroll
    for (int r = 0; r < 8; r++) {
      size_t tok = tokbase + q0 + r + 8 * half;
      int cc = h * HD + dt * 16 + lr;
      out[tok * (2 * AH) + cc] = o[dt][r] / l_r[r];
    }
}

// ---------- launcher ----------
extern "C" void kernel_launch(void* const* d_in, const int* in_sizes, int n_in,
                              void* d_out, int out_size, void* d_ws, size_t ws_size,
                              hipStream_t stream) {
  const float* hidden = (const float*)d_in[0];
  const float* mask   = (const float*)d_in[1];
  const float* Wq = (const float*)d_in[2];  const float* bq = (const float*)d_in[3];
  const float* Wk = (const float*)d_in[4];  const float* bk = (const float*)d_in[5];
  const float* Wv = (const float*)d_in[6];  const float* bv = (const float*)d_in[7];
  const float* dwk = (const float*)d_in[8];
  const float* pwk = (const float*)d_in[9];
  const float* cbias = (const float*)d_in[10];
  const float* Wck = (const float*)d_in[11]; const float* bck = (const float*)d_in[12];
  const float* Wco = (const float*)d_in[13]; const float* bco = (const float*)d_in[14];
  float* out = (float*)d_out;

  char* w = (char*)d_ws;
  size_t off = 0;
  auto alloc = [&](size_t bytes) -> char* {
    char* p = w + off;
    off = (off + bytes + 255) & ~(size_t)255;
    return p;
  };
  bf16_t* Xb   = (bf16_t*)alloc((size_t)NTOK * HIDDEN * 2);  // reused as dw output later
  bf16_t* WqT  = (bf16_t*)alloc((size_t)AH * HIDDEN * 2);
  bf16_t* WkT  = (bf16_t*)alloc((size_t)AH * HIDDEN * 2);
  bf16_t* WvT  = (bf16_t*)alloc((size_t)AH * HIDDEN * 2);
  bf16_t* WcoT = (bf16_t*)alloc((size_t)AH * HIDDEN * 2);
  bf16_t* pwb  = (bf16_t*)alloc((size_t)AH * HIDDEN * 2);
  float* qf  = (float*)alloc((size_t)NTOK * AH * 4);
  float* kf  = (float*)alloc((size_t)NTOK * AH * 4);
  float* vf  = (float*)alloc((size_t)NTOK * AH * 4);
  float* cof = (float*)alloc((size_t)NTOK * AH * 4);
  float* kcf = (float*)alloc((size_t)NTOK * AH * 4);  // key_conv, then conv_attn in place
  bf16_t* qbb = (bf16_t*)alloc((size_t)NTOK * AH * 2);
  bf16_t* kbb = (bf16_t*)alloc((size_t)NTOK * AH * 2);
  bf16_t* vbb = (bf16_t*)alloc((size_t)NTOK * AH * 2);
  float* kernb = (float*)alloc((size_t)NTOK * NH * KW * 4);
  (void)ws_size; (void)in_sizes; (void)n_in; (void)out_size;

  const int T = 256;
  // 1) pack activations + weights to bf16
  cvt_f32_bf16<<<(NTOK * HIDDEN + T - 1) / T, T, 0, stream>>>(hidden, Xb, NTOK * HIDDEN);
  tpack<<<(HIDDEN * AH + T - 1) / T, T, 0, stream>>>(Wq, WqT, HIDDEN, AH);
  tpack<<<(HIDDEN * AH + T - 1) / T, T, 0, stream>>>(Wk, WkT, HIDDEN, AH);
  tpack<<<(HIDDEN * AH + T - 1) / T, T, 0, stream>>>(Wv, WvT, HIDDEN, AH);
  tpack<<<(HIDDEN * AH + T - 1) / T, T, 0, stream>>>(Wco, WcoT, HIDDEN, AH);
  // pw_kernel is already [O, C] = [N][K]; flat convert
  cvt_f32_bf16<<<(HIDDEN * AH + T - 1) / T, T, 0, stream>>>(pwk, pwb, HIDDEN * AH);

  // 2) projection GEMMs (fp32 out + bf16 copies for attention)
  dim3 gg(AH / 128, NTOK / 128);
  gemm_bf16<true ><<<gg, 256, 0, stream>>>(Xb, WqT,  bq,  qf,  qbb, NTOK, AH, HIDDEN);
  gemm_bf16<true ><<<gg, 256, 0, stream>>>(Xb, WkT,  bk,  kf,  kbb, NTOK, AH, HIDDEN);
  gemm_bf16<true ><<<gg, 256, 0, stream>>>(Xb, WvT,  bv,  vf,  vbb, NTOK, AH, HIDDEN);
  gemm_bf16<false><<<gg, 256, 0, stream>>>(Xb, WcoT, bco, cof, (bf16_t*)nullptr, NTOK, AH, HIDDEN);

  // 3) depthwise conv (overwrites Xb — projections above are done with it) + pointwise GEMM
  dwconv<<<(NTOK * HIDDEN + T - 1) / T, T, 0, stream>>>(hidden, dwk, Xb);
  gemm_bf16<false><<<gg, 256, 0, stream>>>(Xb, pwb, cbias, kcf, (bf16_t*)nullptr, NTOK, AH, HIDDEN);

  // 4) conv_attn = key_conv * q (in place), dynamic kernel softmax, windowed contraction
  mul_inplace<<<(NTOK * AH + T - 1) / T, T, 0, stream>>>(kcf, qf, NTOK * AH);
  dynk<<<(NTOK * NH + T - 1) / T, T, 0, stream>>>(kcf, Wck, bck, kernb, NTOK * NH);
  convctx<<<(NTOK * AH + T - 1) / T, T, 0, stream>>>(cof, kernb, out);

  // 5) flash attention -> output cols [0, 384)
  attn<<<BATCH * NH * (SEQ / 64), 128, 0, stream>>>(qbb, kbb, vbb, mask, out);
}
